// LSTMModel_24550033064454
// MI455X (gfx1250) — matrix operand order repaired and checked by reading, not compiled, not run on hardware
//
#include <hip/hip_runtime.h>
#include <hip/hip_bf16.h>
#include <stdint.h>

// Problem dims (fixed by the reference)
#define B_SZ 256
#define T_SZ 512
#define I_SZ 256
#define H_SZ 1024
#define G4   4096           // 4*H
#define NBLK 128            // persistent grid: 2 (M) x 64 (N) blocks
#define NCHUNK 20           // K chunks of 64: 16 from h (K=1024) + 4 from x (K=256)

// Flip to 0 if the async-to-LDS path misbehaves (falls back to load + ds_store).
#define USE_ASYNC_LDS 1

typedef __attribute__((ext_vector_type(16))) __bf16 v16bf;
typedef __attribute__((ext_vector_type(8)))  float  v8f;

union FragB16 { v16bf v; uint4 q[2]; };

__device__ __forceinline__ float fast_sigmoid(float x) { return 1.0f / (1.0f + __expf(-x)); }
__device__ __forceinline__ float fast_tanh(float x)    { return 2.0f * fast_sigmoid(2.0f * x) - 1.0f; }

// ---------------------------------------------------------------------------
// Prep: weights -> bf16 (row-major [4H, K], contiguous-K per output column),
// fused bias, h0 -> bf16, reset the grid-barrier counter (every call, so
// graph replays stay deterministic).
// ---------------------------------------------------------------------------
__global__ void prep_kernel(const float* __restrict__ Whh, const float* __restrict__ Wih,
                            const float* __restrict__ b_ih, const float* __restrict__ b_hh,
                            const float* __restrict__ h0,
                            __bf16* __restrict__ Whh_b, __bf16* __restrict__ Wih_b,
                            float* __restrict__ bias, __bf16* __restrict__ hbuf,
                            int* __restrict__ bar)
{
    const int n_whh = G4 * H_SZ;
    const int n_wih = G4 * I_SZ;
    const int n_h   = B_SZ * H_SZ;
    const int stride = gridDim.x * blockDim.x;
    int idx0 = blockIdx.x * blockDim.x + threadIdx.x;
    if (idx0 == 0) *bar = 0;
    for (int idx = idx0; idx < n_whh; idx += stride) {
        Whh_b[idx] = (__bf16)Whh[idx];
        if (idx < n_wih) Wih_b[idx] = (__bf16)Wih[idx];
        if (idx < G4)    bias[idx]  = b_ih[idx] + b_hh[idx];
        if (idx < n_h)   hbuf[idx]  = (__bf16)h0[idx];
    }
}

// ---------------------------------------------------------------------------
// Persistent LSTM kernel. One launch runs all T=512 steps with a device-wide
// barrier between steps.
//
// Grid: 128 blocks x 256 threads (8 waves).
//   block -> 128 batch rows x 16 hidden cols (x 4 gate slices)
//   wave  -> 16 batch rows; owns 16x16 accumulator tiles for gates i,f,g,o
// W chunks (4 gates x 16 cols x 64 K, bf16 = 8 KB) staged to LDS with
// async-to-LDS loads, triple buffered (one __syncthreads per chunk);
// A (h rows) double buffered in VGPRs. 8 WMMAs per sync window.
// Cell state c is held in registers for the whole sequence.
// ---------------------------------------------------------------------------
__global__ void __launch_bounds__(256)
lstm_persistent(const __bf16* __restrict__ Whh,   // [4H, H]  bf16
                const __bf16* __restrict__ Wih,   // [4H, I]  bf16
                const float*  __restrict__ bias,  // [4H]
                const float*  __restrict__ x,     // [B, T, I] f32
                const float*  __restrict__ c0,    // [1, B, H] f32
                __bf16* __restrict__ hA,          // [B, H] bf16 (holds h0 at entry)
                __bf16* __restrict__ hB,          // [B, H] bf16
                int* __restrict__ bar)
{
    __shared__ __bf16 Bs[3][64][72];   // 3 bufs x (4 gates x 16 cols) x (64 K + 8 pad)

    const int tid    = threadIdx.x;
    const int lane   = tid & 31;
    const int wave   = tid >> 5;          // 0..7
    const int laneHi = lane >> 4;         // 0 | 1
    const int nl     = lane & 15;

    const int nblk   = blockIdx.x & 63;   // hidden-column block
    const int mblk   = blockIdx.x >> 6;   // batch-row block (0|1)
    const int n0     = nblk * 16;
    const int m_base = mblk * 128 + wave * 16;

    const int arow = m_base + nl;         // A-fragment row for this lane
    const int n    = n0 + nl;             // output column for this lane

    // Staging role: each thread moves 32 B of one (gate,col) row per chunk.
    const int s_row  = tid >> 2;          // 0..63 = gate*16 + col
    const int s_gate = s_row >> 4;
    const int s_col  = n0 + (s_row & 15);
    const int s_koff = (tid & 3) * 16;    // element offset within 64-wide chunk

    // Biases are loop-invariant across time.
    const float bi = bias[n];
    const float bf = bias[H_SZ + n];
    const float bg = bias[2 * H_SZ + n];
    const float bo = bias[3 * H_SZ + n];

    // Cell state in registers for the entire sequence.
    float creg[8];
    #pragma unroll
    for (int j = 0; j < 8; ++j)
        creg[j] = c0[(size_t)(m_base + j + laneHi * 8) * H_SZ + n];

    for (int t = 0; t < T_SZ; ++t) {
        const __bf16* hin  = (t & 1) ? hB : hA;
        __bf16*       hout = (t & 1) ? hA : hB;

        v8f accs[4];
        #pragma unroll
        for (int g = 0; g < 4; ++g) accs[g] = (v8f){};

        // ---- stage one 64-wide K chunk of W into LDS (2 x b128 / thread) ----
        auto stage_B = [&](int chunk) {
            const __bf16* src;
            if (chunk < 16)
                src = Whh + ((size_t)(s_gate * H_SZ) + s_col) * H_SZ + chunk * 64 + s_koff;
            else
                src = Wih + ((size_t)(s_gate * H_SZ) + s_col) * I_SZ + (chunk - 16) * 64 + s_koff;
            const int buf = chunk % 3;
            __bf16* dst = &Bs[buf][s_row][s_koff];
#if USE_ASYNC_LDS
            // Bs assumed at LDS offset 0 (only shared object in this kernel).
            // dst is passed as a template-unused operand so Bs's address
            // escapes; with the memory clobber this keeps the ds reads alive.
            const uint32_t lds_off =
                (uint32_t)(((buf * 64 + s_row) * 72 + s_koff) * 2);
            asm volatile("global_load_async_to_lds_b128 %0, %1, off"
                         :: "v"(lds_off), "v"(src), "v"(dst) : "memory");
            asm volatile("global_load_async_to_lds_b128 %0, %1, off"
                         :: "v"(lds_off + 16), "v"(src + 8), "v"(dst) : "memory");
#else
            *(uint4*)dst       = *(const uint4*)src;
            *(uint4*)(dst + 8) = *(const uint4*)(src + 8);
#endif
        };

        auto cvtA = [&](const float* p, FragB16& a) {
            float4 f0 = *(const float4*)(p);
            float4 f1 = *(const float4*)(p + 4);
            float4 f2 = *(const float4*)(p + 16);
            float4 f3 = *(const float4*)(p + 20);
            a.v = (v16bf){ (__bf16)f0.x, (__bf16)f0.y, (__bf16)f0.z, (__bf16)f0.w,
                           (__bf16)f1.x, (__bf16)f1.y, (__bf16)f1.z, (__bf16)f1.w,
                           (__bf16)f2.x, (__bf16)f2.y, (__bf16)f2.z, (__bf16)f2.w,
                           (__bf16)f3.x, (__bf16)f3.y, (__bf16)f3.z, (__bf16)f3.w };
        };

        // A fragments for one 64-wide chunk: two 16x32 sub-fragments.
        auto load_A = [&](int chunk, FragB16& a0, FragB16& a1) {
            if (chunk < 16) {
                const __bf16* p = hin + (size_t)arow * H_SZ + chunk * 64 + laneHi * 8;
                a0.q[0] = *(const uint4*)(p);
                a0.q[1] = *(const uint4*)(p + 16);
                a1.q[0] = *(const uint4*)(p + 32);
                a1.q[1] = *(const uint4*)(p + 48);
            } else {
                const float* p = x + ((size_t)arow * T_SZ + t) * I_SZ
                                   + (chunk - 16) * 64 + laneHi * 8;
                cvtA(p, a0);
                cvtA(p + 32, a1);
            }
        };

        // ---- pipelined K loop ----
        stage_B(0);
        FragB16 a0c, a1c, a0n, a1n;
        load_A(0, a0c, a1c);

        for (int chunk = 0; chunk < NCHUNK; ++chunk) {
            if (chunk + 1 < NCHUNK) {
                stage_B(chunk + 1);                 // into buf (chunk+1)%3
#if USE_ASYNC_LDS
                asm volatile("s_wait_asynccnt 0x2" ::: "memory");  // chunk's 2 ops done
#endif
            } else {
#if USE_ASYNC_LDS
                asm volatile("s_wait_asynccnt 0x0" ::: "memory");
#endif
            }
            __syncthreads();                        // chunk's buffer fully staged
            if (chunk + 1 < NCHUNK) load_A(chunk + 1, a0n, a1n);   // overlap w/ WMMAs

            const int buf = chunk % 3;
            #pragma unroll
            for (int s = 0; s < 2; ++s) {
                const v16bf av = s ? a1c.v : a0c.v;
                // Load all four gate fragments into distinct registers first
                // so the ds waits are incremental, not one full wait per WMMA.
                FragB16 b0, b1, b2, b3;
                const int ko = s * 32 + laneHi * 8;
                const __bf16* p0 = &Bs[buf][ 0 + nl][ko];
                const __bf16* p1 = &Bs[buf][16 + nl][ko];
                const __bf16* p2 = &Bs[buf][32 + nl][ko];
                const __bf16* p3 = &Bs[buf][48 + nl][ko];
                b0.q[0] = *(const uint4*)(p0);  b0.q[1] = *(const uint4*)(p0 + 16);
                b1.q[0] = *(const uint4*)(p1);  b1.q[1] = *(const uint4*)(p1 + 16);
                b2.q[0] = *(const uint4*)(p2);  b2.q[1] = *(const uint4*)(p2 + 16);
                b3.q[0] = *(const uint4*)(p3);  b3.q[1] = *(const uint4*)(p3 + 16);
                accs[0] = __builtin_amdgcn_wmma_f32_16x16x32_bf16(
                              false, av, false, b0.v, (short)0, accs[0], false, false);
                accs[1] = __builtin_amdgcn_wmma_f32_16x16x32_bf16(
                              false, av, false, b1.v, (short)0, accs[1], false, false);
                accs[2] = __builtin_amdgcn_wmma_f32_16x16x32_bf16(
                              false, av, false, b2.v, (short)0, accs[2], false, false);
                accs[3] = __builtin_amdgcn_wmma_f32_16x16x32_bf16(
                              false, av, false, b3.v, (short)0, accs[3], false, false);
            }
            a0c = a0n;
            a1c = a1n;
        }

        // ---- fused gate epilogue (c in registers) ----
        #pragma unroll
        for (int j = 0; j < 8; ++j) {
            const float i_ = fast_sigmoid(accs[0][j] + bi);
            const float f_ = fast_sigmoid(accs[1][j] + bf);
            const float g_ = fast_tanh   (accs[2][j] + bg);
            const float o_ = fast_sigmoid(accs[3][j] + bo);
            creg[j] = f_ * creg[j] + i_ * g_;
            hout[(size_t)(m_base + j + laneHi * 8) * H_SZ + n] =
                (__bf16)(o_ * fast_tanh(creg[j]));
        }

        // ---- device-wide barrier between timesteps ----
        if (t != T_SZ - 1) {
            __builtin_amdgcn_fence(__ATOMIC_RELEASE, "agent");
            __syncthreads();
            if (tid == 0) {
                __hip_atomic_fetch_add(bar, 1, __ATOMIC_RELAXED, __HIP_MEMORY_SCOPE_AGENT);
                const int target = NBLK * (t + 1);
                while (__hip_atomic_load(bar, __ATOMIC_RELAXED, __HIP_MEMORY_SCOPE_AGENT) < target)
                    __builtin_amdgcn_s_sleep(1);
            }
            __syncthreads();
            __builtin_amdgcn_fence(__ATOMIC_ACQUIRE, "agent");
        }
    }
}

// ---------------------------------------------------------------------------
// Decoder (tiny): y1 = h W1^T + b1 ; y = y1 W2^T + b2
// ---------------------------------------------------------------------------
__global__ void dec1_kernel(const __bf16* __restrict__ h, const float* __restrict__ W1,
                            const float* __restrict__ b1, float* __restrict__ y1)
{
    const int idx = blockIdx.x * blockDim.x + threadIdx.x;  // B * 512 threads
    const int b = idx >> 9, j = idx & 511;
    const __bf16* hp = h + (size_t)b * H_SZ;
    const float*  wp = W1 + (size_t)j * H_SZ;
    float s = b1[j];
    #pragma unroll 4
    for (int k = 0; k < H_SZ; ++k) s += (float)hp[k] * wp[k];
    y1[idx] = s;
}

__global__ void dec2_kernel(const float* __restrict__ y1, const float* __restrict__ W2,
                            const float* __restrict__ b2, float* __restrict__ out)
{
    const int idx = blockIdx.x * blockDim.x + threadIdx.x;  // B * 4 threads
    const int b = idx >> 2, o = idx & 3;
    const float* yp = y1 + (size_t)b * 512;
    const float* wp = W2 + (size_t)o * 512;
    float s = b2[o];
    #pragma unroll 4
    for (int k = 0; k < 512; ++k) s += yp[k] * wp[k];
    out[idx] = s;
}

// ---------------------------------------------------------------------------
extern "C" void kernel_launch(void* const* d_in, const int* in_sizes, int n_in,
                              void* d_out, int out_size, void* d_ws, size_t ws_size,
                              hipStream_t stream)
{
    const float* x    = (const float*)d_in[0];
    const float* h0   = (const float*)d_in[1];
    const float* c0   = (const float*)d_in[2];
    const float* W_ih = (const float*)d_in[3];
    const float* W_hh = (const float*)d_in[4];
    const float* b_ih = (const float*)d_in[5];
    const float* b_hh = (const float*)d_in[6];
    const float* W1   = (const float*)d_in[7];
    const float* b1   = (const float*)d_in[8];
    const float* W2   = (const float*)d_in[9];
    const float* b2   = (const float*)d_in[10];
    float* out = (float*)d_out;

    // Workspace carve-out (~11.6 MB)
    char* ws = (char*)d_ws;
    __bf16* Whh_b = (__bf16*)ws;  ws += (size_t)G4 * H_SZ * 2;    // 8 MB
    __bf16* Wih_b = (__bf16*)ws;  ws += (size_t)G4 * I_SZ * 2;    // 2 MB
    float*  bias  = (float*)ws;   ws += (size_t)G4 * 4;           // 16 KB
    __bf16* hA    = (__bf16*)ws;  ws += (size_t)B_SZ * H_SZ * 2;  // 512 KB
    __bf16* hB    = (__bf16*)ws;  ws += (size_t)B_SZ * H_SZ * 2;  // 512 KB
    float*  y1    = (float*)ws;   ws += (size_t)B_SZ * 512 * 4;   // 512 KB
    int*    bar   = (int*)ws;     ws += 256;

    prep_kernel<<<2048, 256, 0, stream>>>(W_hh, W_ih, b_ih, b_hh, h0,
                                          Whh_b, Wih_b, bias, hA, bar);

    lstm_persistent<<<NBLK, 256, 0, stream>>>(Whh_b, Wih_b, bias, x, c0, hA, hB, bar);

    // T=512 is even, so the final hidden state lands back in hA.
    dec1_kernel<<<(B_SZ * 512) / 256, 256, 0, stream>>>(hA, W1, b1, y1);
    dec2_kernel<<<(B_SZ * 4 + 255) / 256, 256, 0, stream>>>(y1, W2, b2, out);
}
